// Seq2Seq_34591666602628
// MI455X (gfx1250) — 16-bit and 8-bit matrix operand order repaired and checked by reading, compiled, not run on hardware
//
#include <hip/hip_runtime.h>
#include <hip/hip_bf16.h>
#include <math.h>

typedef __attribute__((ext_vector_type(16))) _Float16 v16h;
typedef __attribute__((ext_vector_type(8)))  float    v8f;
typedef __attribute__((ext_vector_type(16))) int      v16i;

union HFrag { v16h v; float4 f4[2]; };

#define LS 48
#define LT 48
#define BB 64
#define EE 512
#define HH 512

__global__ void k_zero_f32(float* __restrict__ d, int n) {
    for (int i = blockIdx.x * blockDim.x + threadIdx.x; i < n; i += gridDim.x * blockDim.x)
        d[i] = 0.f;
}

__global__ void k_cvt_f16(const float* __restrict__ s, _Float16* __restrict__ d,
                          int sld, int dld, int rows, int cols) {
    int n = rows * cols;
    for (int i = blockIdx.x * blockDim.x + threadIdx.x; i < n; i += gridDim.x * blockDim.x) {
        int r = i / cols, c = i - r * cols;
        d[(size_t)r * dld + c] = (_Float16)s[(size_t)r * sld + c];
    }
}

__global__ void k_addv(const float* __restrict__ a, const float* __restrict__ b,
                       float* __restrict__ o, int n) {
    for (int i = blockIdx.x * blockDim.x + threadIdx.x; i < n; i += gridDim.x * blockDim.x)
        o[i] = a[i] + b[i];
}

__device__ inline unsigned char cvt_e4m3(float f) {
    unsigned char s = (f < 0.f) ? 0x80 : 0x00;
    float a = fabsf(f);
    if (!(a == a)) return 0x7E;
    if (a >= 448.f) return s | 0x7E;
    if (a < 0.015625f) {
        int q = (int)rintf(a * 512.f);
        if (q >= 8) return s | 0x08;
        return s | (unsigned char)q;
    }
    int e; float m = frexpf(a, &e);
    int q = (int)rintf(m * 16.f);
    int E = e + 6;
    if (q == 16) { q = 8; E += 1; }
    if (E > 15) return s | 0x7E;
    return s | (unsigned char)((E << 3) | (q & 7));
}

__global__ void k_cvt_fp8(const float* __restrict__ s, unsigned char* __restrict__ d,
                          long long n, float scale) {
    for (long long i = blockIdx.x * blockDim.x + threadIdx.x; i < n;
         i += (long long)gridDim.x * blockDim.x)
        d[i] = cvt_e4m3(s[i] * scale);
}

__global__ void k_cvt_h_fp8(const _Float16* __restrict__ s, unsigned char* __restrict__ d,
                            int n, float scale) {
    for (int i = blockIdx.x * blockDim.x + threadIdx.x; i < n; i += gridDim.x * blockDim.x)
        d[i] = cvt_e4m3((float)s[i] * scale);
}

__global__ void k_gather_emb(const int* __restrict__ idx, const float* __restrict__ emb,
                             _Float16* __restrict__ out, int ldo, int n_tok) {
    int tkn = blockIdx.x;
    if (tkn >= n_tok) return;
    int row = idx[tkn];
    const float* s = emb + (size_t)row * EE;
    _Float16* d = out + (size_t)tkn * ldo;
    for (int i = threadIdx.x; i < EE; i += blockDim.x) d[i] = (_Float16)s[i];
}

__global__ void k_gemm_xWT(const _Float16* __restrict__ A, int lda,
                           const _Float16* __restrict__ W, int ldw,
                           const float* __restrict__ bias,
                           float* __restrict__ Cf, int ldc,
                           _Float16* __restrict__ Ch, int ldch,
                           int K, int act) {
    const int tid   = threadIdx.x;
    const int wave  = tid >> 5;
    const int lane  = tid & 31;
    const int lrow  = lane & 15;
    const int khalf = lane >> 4;
    const int n0 = blockIdx.x * 16;
    const int m0 = blockIdx.y * 64 + wave * 16;

    const _Float16* Arow = A + (size_t)(m0 + lrow) * lda;
    const _Float16* Wrow = W + (size_t)(n0 + lrow) * ldw;

    v8f acc = {0.f,0.f,0.f,0.f,0.f,0.f,0.f,0.f};
    for (int k = 0; k < K; k += 32) {
        HFrag af, bf;
        af.f4[0] = *(const float4*)(Arow + k + khalf * 8);
        af.f4[1] = *(const float4*)(Arow + k + 16 + khalf * 8);
        bf.f4[0] = *(const float4*)(Wrow + k + khalf * 8);
        bf.f4[1] = *(const float4*)(Wrow + k + 16 + khalf * 8);
        acc = __builtin_amdgcn_wmma_f32_16x16x32_f16(false, af.v, false, bf.v,
                                                     (short)0, acc, false, false);
    }
    const int col = n0 + lrow;
    const float bv = bias ? bias[col] : 0.f;
#pragma unroll
    for (int i = 0; i < 8; i++) {
        int row = m0 + khalf * 8 + i;
        float v = acc[i] + bv;
        if (act == 1) v = tanhf(v);
        if (Cf) Cf[(size_t)row * ldc + col] = v;
        if (Ch) Ch[(size_t)row * ldch + col] = (_Float16)v;
    }
}

__global__ void k_gemm_fp8(const unsigned char* __restrict__ A, int lda,
                           const unsigned char* __restrict__ W, int ldw,
                           const float* __restrict__ bias,
                           float* __restrict__ Cf, int ldc,
                           int K, float oscale) {
    const int tid   = threadIdx.x;
    const int wave  = tid >> 5;
    const int lane  = tid & 31;
    const int lrow  = lane & 15;
    const int khalf = lane >> 4;
    const int n0 = blockIdx.x * 16;
    const int m0 = wave * 16;

    const unsigned char* Arow = A + (size_t)(m0 + lrow) * lda;
    const unsigned char* Wrow = W + (size_t)(n0 + lrow) * ldw;

    v8f acc = {0.f,0.f,0.f,0.f,0.f,0.f,0.f,0.f};
    for (int k = 0; k < K; k += 128) {
        __builtin_prefetch(Wrow + k + 128, 0, 3);
        union { v16i v; unsigned long long q[8]; } af;
        union { v16i v; float4 f4[4]; } bf;
#pragma unroll
        for (int i = 0; i < 8; i++)
            af.q[i] = *(const unsigned long long*)(Arow + k + 32 * (i >> 1) + khalf * 16 + 8 * (i & 1));
#pragma unroll
        for (int q = 0; q < 4; q++)
            bf.f4[q] = *(const float4*)(Wrow + k + q * 32 + khalf * 16);
        acc = __builtin_amdgcn_wmma_f32_16x16x128_fp8_fp8(af.v, bf.v, (short)0, acc,
                                                          false, false);
    }
    const int col = n0 + lrow;
    const float bv = bias ? bias[col] : 0.f;
#pragma unroll
    for (int i = 0; i < 8; i++) {
        int row = m0 + khalf * 8 + i;
        Cf[(size_t)row * ldc + col] = acc[i] * oscale + bv;
    }
}

__global__ void k_lstm_step(const _Float16* __restrict__ X, int Kx,
                            const _Float16* __restrict__ Wih,
                            const _Float16* __restrict__ Whh,
                            const float* __restrict__ bsum,
                            const _Float16* __restrict__ Hprev,
                            float* __restrict__ Cst,
                            _Float16* __restrict__ Hnext,
                            _Float16* __restrict__ Hseq, int seq_ld) {
    const int tid   = threadIdx.x;
    const int wave  = tid >> 5;
    const int lane  = tid & 31;
    const int lrow  = lane & 15;
    const int khalf = lane >> 4;
    const int j0 = blockIdx.x * 16;

    const _Float16* Wihrow = Wih + (size_t)(wave * HH + j0 + lrow) * Kx;
    const _Float16* Whhrow = Whh + (size_t)(wave * HH + j0 + lrow) * HH;

    v8f acc[4];
#pragma unroll
    for (int ms = 0; ms < 4; ms++) acc[ms] = (v8f){0.f,0.f,0.f,0.f,0.f,0.f,0.f,0.f};

    for (int k = 0; k < Kx; k += 32) {
        HFrag bf;
        bf.f4[0] = *(const float4*)(Wihrow + k + khalf * 8);
        bf.f4[1] = *(const float4*)(Wihrow + k + 16 + khalf * 8);
#pragma unroll
        for (int ms = 0; ms < 4; ms++) {
            const _Float16* Arow = X + (size_t)(ms * 16 + lrow) * Kx;
            HFrag af;
            af.f4[0] = *(const float4*)(Arow + k + khalf * 8);
            af.f4[1] = *(const float4*)(Arow + k + 16 + khalf * 8);
            acc[ms] = __builtin_amdgcn_wmma_f32_16x16x32_f16(false, af.v, false, bf.v,
                                                             (short)0, acc[ms], false, false);
        }
    }
    for (int k = 0; k < HH; k += 32) {
        HFrag bf;
        bf.f4[0] = *(const float4*)(Whhrow + k + khalf * 8);
        bf.f4[1] = *(const float4*)(Whhrow + k + 16 + khalf * 8);
#pragma unroll
        for (int ms = 0; ms < 4; ms++) {
            const _Float16* Arow = Hprev + (size_t)(ms * 16 + lrow) * HH;
            HFrag af;
            af.f4[0] = *(const float4*)(Arow + k + khalf * 8);
            af.f4[1] = *(const float4*)(Arow + k + 16 + khalf * 8);
            acc[ms] = __builtin_amdgcn_wmma_f32_16x16x32_f16(false, af.v, false, bf.v,
                                                             (short)0, acc[ms], false, false);
        }
    }

    __shared__ float g4[4][64][16];
#pragma unroll
    for (int ms = 0; ms < 4; ms++)
#pragma unroll
        for (int i = 0; i < 8; i++)
            g4[wave][ms * 16 + khalf * 8 + i][lrow] = acc[ms][i];
    __syncthreads();

    for (int e = tid; e < 64 * 16; e += 128) {
        int m = e >> 4, j = e & 15;
        int jg = j0 + j;
        float gi = g4[0][m][j] + bsum[0 * HH + jg];
        float gf = g4[1][m][j] + bsum[1 * HH + jg];
        float gg = g4[2][m][j] + bsum[2 * HH + jg];
        float go = g4[3][m][j] + bsum[3 * HH + jg];
        float si = 1.f / (1.f + __expf(-gi));
        float sf = 1.f / (1.f + __expf(-gf));
        float so = 1.f / (1.f + __expf(-go));
        float c = sf * Cst[(size_t)m * HH + jg] + si * tanhf(gg);
        float h = so * tanhf(c);
        Cst[(size_t)m * HH + jg] = c;
        Hnext[(size_t)m * HH + jg] = (_Float16)h;
        if (Hseq) Hseq[(size_t)m * seq_ld + jg] = (_Float16)h;
    }
}

__global__ void k_attn_score(const float* __restrict__ hs_proj, const float* __restrict__ q,
                             const float* __restrict__ v, float* __restrict__ score) {
    int l = blockIdx.x, b = blockIdx.y, tid = threadIdx.x;
    const float* hp = hs_proj + ((size_t)l * BB + b) * 1024;
    const float* qb = q + (size_t)b * 1024;
    float s = 0.f;
    for (int i = tid; i < 1024; i += 256) s += tanhf(hp[i] + qb[i]) * v[i];
    __shared__ float red[256];
    red[tid] = s; __syncthreads();
    for (int st = 128; st > 0; st >>= 1) {
        if (tid < st) red[tid] += red[tid + st];
        __syncthreads();
    }
    if (tid == 0) score[b * LS + l] = red[0];
}

__global__ void k_softmax48(const float* __restrict__ score, float* __restrict__ ps) {
    int b = blockIdx.x, tid = threadIdx.x;
    __shared__ float s[LS];
    __shared__ float mx, sum;
    if (tid < LS) s[tid] = score[b * LS + tid];
    __syncthreads();
    if (tid == 0) {
        float m = s[0];
        for (int i = 1; i < LS; i++) m = fmaxf(m, s[i]);
        mx = m;
    }
    __syncthreads();
    if (tid < LS) s[tid] = __expf(s[tid] - mx);
    __syncthreads();
    if (tid == 0) {
        float t = 0.f;
        for (int i = 0; i < LS; i++) t += s[i];
        sum = t;
    }
    __syncthreads();
    if (tid < LS) ps[b * LS + tid] = s[tid] / sum;
}

__global__ void k_attn_ctx(const float* __restrict__ ps, const _Float16* __restrict__ hs,
                           const _Float16* __restrict__ hcur, _Float16* __restrict__ lin_in) {
    int b = blockIdx.x, tid = threadIdx.x;
    __shared__ float sp[LS];
    if (tid < LS) sp[tid] = ps[b * LS + tid];
    __syncthreads();
    for (int k = tid; k < 1024; k += blockDim.x) {
        float a = 0.f;
        for (int l = 0; l < LS; l++) a += sp[l] * (float)hs[((size_t)l * BB + b) * 1024 + k];
        lin_in[(size_t)b * 1536 + k] = (_Float16)a;
    }
    for (int j = tid; j < HH; j += blockDim.x)
        lin_in[(size_t)b * 1536 + 1024 + j] = hcur[(size_t)b * HH + j];
}

__global__ void k_nll(const float* __restrict__ logits, const int* __restrict__ ts,
                      int tstep, float* __restrict__ acc2) {
    int b = blockIdx.x, tid = threadIdx.x;
    const float* row = logits + (size_t)b * 32000;
    __shared__ float red[256];
    float mx = -1e30f;
    for (int i = tid; i < 32000; i += 256) mx = fmaxf(mx, row[i]);
    red[tid] = mx; __syncthreads();
    for (int st = 128; st > 0; st >>= 1) {
        if (tid < st) red[tid] = fmaxf(red[tid], red[tid + st]);
        __syncthreads();
    }
    mx = red[0]; __syncthreads();
    float s = 0.f;
    for (int i = tid; i < 32000; i += 256) s += __expf(row[i] - mx);
    red[tid] = s; __syncthreads();
    for (int st = 128; st > 0; st >>= 1) {
        if (tid < st) red[tid] += red[tid + st];
        __syncthreads();
    }
    if (tid == 0) {
        int tgt = ts[(tstep + 1) * BB + b];
        if (tgt != 0) {
            float lp = row[tgt] - mx - logf(red[0]);
            atomicAdd(&acc2[0], -lp);
            atomicAdd(&acc2[1], 1.0f);
        }
    }
}

__global__ void k_finalize(const float* __restrict__ acc2, float* __restrict__ out) {
    out[0] = acc2[0] / acc2[1];
}

extern "C" void kernel_launch(void* const* d_in, const int* in_sizes, int n_in,
                              void* d_out, int out_size, void* d_ws, size_t ws_size,
                              hipStream_t stream) {
    const int*   xs       = (const int*)  d_in[0];
    const int*   ts       = (const int*)  d_in[1];
    const float* src_emb  = (const float*)d_in[2];
    const float* tgt_emb  = (const float*)d_in[3];
    const float* encf_Wih = (const float*)d_in[4];
    const float* encf_Whh = (const float*)d_in[5];
    const float* encf_bih = (const float*)d_in[6];
    const float* encf_bhh = (const float*)d_in[7];
    const float* encb_Wih = (const float*)d_in[8];
    const float* encb_Whh = (const float*)d_in[9];
    const float* encb_bih = (const float*)d_in[10];
    const float* encb_bhh = (const float*)d_in[11];
    const float* dec_Wih  = (const float*)d_in[12];
    const float* dec_Whh  = (const float*)d_in[13];
    const float* dec_bih  = (const float*)d_in[14];
    const float* dec_bhh  = (const float*)d_in[15];
    const float* attn_W   = (const float*)d_in[16];
    const float* attn_v   = (const float*)d_in[17];
    const float* lin_W    = (const float*)d_in[18];
    const float* lin_b    = (const float*)d_in[19];
    const float* out_W    = (const float*)d_in[20];
    const float* out_b    = (const float*)d_in[21];

    char* p = (char*)d_ws;
    auto alloc = [&](size_t bytes) -> void* {
        void* r = (void*)p;
        p += (bytes + 255) & ~(size_t)255;
        return r;
    };
    _Float16* Wfih_h = (_Float16*)alloc((size_t)2048 * 512 * 2);
    _Float16* Wfhh_h = (_Float16*)alloc((size_t)2048 * 512 * 2);
    _Float16* Wbih_h = (_Float16*)alloc((size_t)2048 * 512 * 2);
    _Float16* Wbhh_h = (_Float16*)alloc((size_t)2048 * 512 * 2);
    _Float16* Wdih_h = (_Float16*)alloc((size_t)2048 * 1024 * 2);
    _Float16* Wdhh_h = (_Float16*)alloc((size_t)2048 * 512 * 2);
    _Float16* Wq_h   = (_Float16*)alloc((size_t)1024 * 512 * 2);
    _Float16* Wk_h   = (_Float16*)alloc((size_t)1024 * 1024 * 2);
    _Float16* WlinH  = (_Float16*)alloc((size_t)512 * 1536 * 2);
    unsigned char* Wout8 = (unsigned char*)alloc((size_t)32000 * 512);
    float*    b_encf = (float*)alloc(2048 * 4);
    float*    b_encb = (float*)alloc(2048 * 4);
    float*    b_dec  = (float*)alloc(2048 * 4);
    _Float16* embs_h = (_Float16*)alloc((size_t)LS * BB * EE * 2);
    _Float16* x_all  = (_Float16*)alloc((size_t)48 * BB * 1024 * 2);
    _Float16* hs_h   = (_Float16*)alloc((size_t)LS * BB * 1024 * 2);
    float*    hs_pr  = (float*)alloc((size_t)LS * BB * 1024 * 4);
    _Float16* h_enc  = (_Float16*)alloc((size_t)2 * 2 * BB * HH * 2);
    float*    c_enc  = (float*)alloc((size_t)2 * BB * HH * 4);
    _Float16* h_dec  = (_Float16*)alloc((size_t)2 * BB * HH * 2);
    float*    c_dec  = (float*)alloc((size_t)BB * HH * 4);
    unsigned char* h8 = (unsigned char*)alloc((size_t)BB * HH);
    float*    qbuf   = (float*)alloc((size_t)BB * 1024 * 4);
    float*    score  = (float*)alloc((size_t)BB * LS * 4);
    float*    psbuf  = (float*)alloc((size_t)BB * LS * 4);
    _Float16* lin_in = (_Float16*)alloc((size_t)BB * 1536 * 2);
    float*    logits = (float*)alloc((size_t)BB * 32000 * 4);
    float*    acc2   = (float*)alloc(2 * 4);

    auto gs = [](long long n) -> int {
        long long g = (n + 255) / 256;
        return (int)(g > 2048 ? 2048 : g);
    };

    k_zero_f32<<<1, 64, 0, stream>>>(acc2, 2);
    k_zero_f32<<<gs(2*2*BB*HH/2), 256, 0, stream>>>((float*)h_enc, 2*2*BB*HH/2);
    k_zero_f32<<<gs(2*BB*HH),     256, 0, stream>>>(c_enc, 2*BB*HH);
    k_zero_f32<<<gs(2*BB*HH/2),   256, 0, stream>>>((float*)h_dec, 2*BB*HH/2);
    k_zero_f32<<<gs(BB*HH),       256, 0, stream>>>(c_dec, BB*HH);
    k_zero_f32<<<gs((long long)48*BB*1024/2), 256, 0, stream>>>((float*)x_all, 48*BB*1024/2);

    k_cvt_f16<<<gs(2048*512), 256, 0, stream>>>(encf_Wih, Wfih_h, 512, 512, 2048, 512);
    k_cvt_f16<<<gs(2048*512), 256, 0, stream>>>(encf_Whh, Wfhh_h, 512, 512, 2048, 512);
    k_cvt_f16<<<gs(2048*512), 256, 0, stream>>>(encb_Wih, Wbih_h, 512, 512, 2048, 512);
    k_cvt_f16<<<gs(2048*512), 256, 0, stream>>>(encb_Whh, Wbhh_h, 512, 512, 2048, 512);
    k_cvt_f16<<<gs(2048*1024), 256, 0, stream>>>(dec_Wih, Wdih_h, 1024, 1024, 2048, 1024);
    k_cvt_f16<<<gs(2048*512), 256, 0, stream>>>(dec_Whh, Wdhh_h, 512, 512, 2048, 512);
    k_cvt_f16<<<gs(1024*512), 256, 0, stream>>>(attn_W, Wq_h, 1536, 512, 1024, 512);
    k_cvt_f16<<<gs(1024*1024), 256, 0, stream>>>(attn_W + 512, Wk_h, 1536, 1024, 1024, 1024);
    k_cvt_f16<<<gs(512*1536), 256, 0, stream>>>(lin_W, WlinH, 1536, 1536, 512, 1536);
    k_cvt_fp8<<<2048, 256, 0, stream>>>(out_W, Wout8, (long long)32000 * 512, 16.f);

    k_addv<<<gs(2048), 256, 0, stream>>>(encf_bih, encf_bhh, b_encf, 2048);
    k_addv<<<gs(2048), 256, 0, stream>>>(encb_bih, encb_bhh, b_encb, 2048);
    k_addv<<<gs(2048), 256, 0, stream>>>(dec_bih, dec_bhh, b_dec, 2048);

    k_gather_emb<<<LS * BB, 128, 0, stream>>>(xs, src_emb, embs_h, 512, LS * BB);
    k_gather_emb<<<(LT - 1) * BB, 128, 0, stream>>>(ts, tgt_emb, x_all, 1024, (LT - 1) * BB);

    for (int t = 0; t < LS; t++) {
        int pp = t & 1, np = pp ^ 1;
        k_lstm_step<<<32, 128, 0, stream>>>(
            embs_h + (size_t)t * BB * EE, 512,
            Wfih_h, Wfhh_h, b_encf,
            h_enc + ((size_t)(0 * 2 + pp)) * BB * HH, c_enc + 0,
            h_enc + ((size_t)(0 * 2 + np)) * BB * HH,
            hs_h + (size_t)t * BB * 1024 + 0, 1024);
        k_lstm_step<<<32, 128, 0, stream>>>(
            embs_h + (size_t)(LS - 1 - t) * BB * EE, 512,
            Wbih_h, Wbhh_h, b_encb,
            h_enc + ((size_t)(1 * 2 + pp)) * BB * HH, c_enc + (size_t)BB * HH,
            h_enc + ((size_t)(1 * 2 + np)) * BB * HH,
            hs_h + (size_t)(LS - 1 - t) * BB * 1024 + 512, 1024);
    }

    k_gemm_xWT<<<dim3(1024 / 16, (LS * BB) / 64), 128, 0, stream>>>(
        hs_h, 1024, Wk_h, 1024, nullptr, hs_pr, 1024, nullptr, 0, 1024, 0);

    for (int t = 0; t < LT - 1; t++) {
        int pp = t & 1, np = pp ^ 1;
        _Float16* hcur = h_dec + (size_t)np * BB * HH;

        k_lstm_step<<<32, 128, 0, stream>>>(
            x_all + (size_t)t * BB * 1024, 1024,
            Wdih_h, Wdhh_h, b_dec,
            h_dec + (size_t)pp * BB * HH, c_dec, hcur,
            nullptr, 0);

        k_gemm_xWT<<<dim3(1024 / 16, 1), 128, 0, stream>>>(
            hcur, 512, Wq_h, 512, nullptr, qbuf, 1024, nullptr, 0, 512, 0);

        k_attn_score<<<dim3(LS, BB), 256, 0, stream>>>(hs_pr, qbuf, attn_v, score);
        k_softmax48<<<BB, 64, 0, stream>>>(score, psbuf);
        k_attn_ctx<<<BB, 256, 0, stream>>>(psbuf, hs_h, hcur, lin_in);

        k_gemm_xWT<<<dim3(512 / 16, 1), 128, 0, stream>>>(
            lin_in, 1536, WlinH, 1536, lin_b,
            nullptr, 0,
            x_all + (size_t)(t + 1) * BB * 1024 + 512, 1024,
            1536, 1);

        k_cvt_h_fp8<<<gs(BB * HH), 256, 0, stream>>>(hcur, h8, BB * HH, 16.f);
        k_gemm_fp8<<<dim3(32000 / 16, 1), 128, 0, stream>>>(
            h8, 512, Wout8, 512, out_b, logits, 32000, 512, 1.f / 256.f);

        k_nll<<<BB, 256, 0, stream>>>(logits, ts, t, acc2);
    }

    k_finalize<<<1, 1, 0, stream>>>(acc2, (float*)d_out);
}
